// MPNet_MoE_17961553232374
// MI455X (gfx1250) — matrix operand order repaired and checked by reading, not compiled, not run on hardware
//
#include <hip/hip_runtime.h>

// ---------------------------------------------------------------------------
// Types for CDNA5 WMMA (wave32)
// ---------------------------------------------------------------------------
typedef __attribute__((ext_vector_type(16))) __bf16 v16bf;
typedef __attribute__((ext_vector_type(8)))  __bf16 v8bf;
typedef __attribute__((ext_vector_type(8)))  float  v8f;

// RNE float -> bf16
__device__ __forceinline__ __bf16 f2bf(float f) {
    union { float f; unsigned int u; } in; in.f = f;
    unsigned int x = in.u;
    unsigned int r = x + 0x7FFFu + ((x >> 16) & 1u);
    union { unsigned short s; __bf16 b; } out;
    out.s = (unsigned short)(r >> 16);
    return out.b;
}

__device__ __forceinline__ v8f wmma_bf16(v16bf a, v16bf b, v8f c) {
    return __builtin_amdgcn_wmma_f32_16x16x32_bf16(
        /*neg_a=*/false, a, /*neg_b=*/false, b,
        /*c_mod=*/(short)0, c, /*reuse_a=*/false, /*reuse_b=*/false);
}

// A fragment: 16x32 (MxK) bf16, A row-major [M][lda].
// ISA layout: lanes 0-15 row M=lane, K={0..7,16..23}; lanes 16-31 row M=lane-16, K=+8.
__device__ __forceinline__ v16bf load_a_bf16(const __bf16* A, int lda, int row,
                                             int k0, int lane) {
    int kb = k0 + ((lane >> 4) << 3);
    const __bf16* p = A + (long)row * lda + kb;
    v8bf lo = *(const v8bf*)(p);        // K = kb..kb+7
    v8bf hi = *(const v8bf*)(p + 16);   // K = kb+16..kb+23
    return __builtin_shufflevector(lo, hi, 0,1,2,3,4,5,6,7,8,9,10,11,12,13,14,15);
}

// B fragment: 32x16 (KxN) bf16 sourced from BT row-major [N][ldb] (BT[n][k]).
// ISA layout: lanes 0-15 col N=lane, K={0..7,16..23}; lanes 16-31 col N=lane-16, K=+8.
__device__ __forceinline__ v16bf load_b_bf16(const __bf16* BT, int ldb, int n0,
                                             int k0, int lane) {
    int n  = n0 + (lane & 15);
    int kb = k0 + ((lane >> 4) << 3);
    const __bf16* p = BT + (long)n * ldb + kb;
    v8bf lo = *(const v8bf*)(p);
    v8bf hi = *(const v8bf*)(p + 16);
    return __builtin_shufflevector(lo, hi, 0,1,2,3,4,5,6,7,8,9,10,11,12,13,14,15);
}

// ---------------------------------------------------------------------------
// Problem constants
// ---------------------------------------------------------------------------
#define B_   128
#define T_   100
#define IN_  512
#define H_   1024
#define G3_  3072   // 3*H
#define HH_  512
#define E_   8
#define M_   (B_ * T_)   // 12800 rows of X / GI

// ---------------------------------------------------------------------------
// Pack kernels
// ---------------------------------------------------------------------------
__global__ void f32_to_bf16_kernel(const float* __restrict__ src,
                                   __bf16* __restrict__ dst, long n) {
    long i = (long)blockIdx.x * blockDim.x + threadIdx.x;
    if (i < n) dst[i] = f2bf(src[i]);
}

// w[e][i][o] f32 -> wT[e][o][i] bf16
__global__ void transpose_moe_w_kernel(const float* __restrict__ w,
                                       __bf16* __restrict__ wT, int I, int O) {
    long idx = (long)blockIdx.x * blockDim.x + threadIdx.x;
    long per = (long)I * O;
    long total = (long)E_ * per;
    if (idx >= total) return;
    int  e  = (int)(idx / per);
    long r  = idx % per;
    int  ii = (int)(r / O);
    int  oo = (int)(r % O);
    wT[((long)e * O + oo) * I + ii] = f2bf(w[idx]);
}

__global__ void zero_h_kernel(float* __restrict__ hf, __bf16* __restrict__ hb) {
    int i = blockIdx.x * 256 + threadIdx.x;
    if (i < B_ * H_) { hf[i] = 0.0f; hb[i] = f2bf(0.0f); }
}

// ---------------------------------------------------------------------------
// GI = X @ mw_ih.T + mb_ih     (M=12800, N=3072, K=512), bf16 WMMA, f32 out
// Block tile 64(M) x 128(N); 8 waves as 2(m) x 4(n); wave tile 32x32 (2x2
// 16x16 tiles -> 4 WMMAs per 4 fragment loads).  grid (200, 24), block 256.
// ---------------------------------------------------------------------------
__global__ void gi_gemm_kernel(const __bf16* __restrict__ Xb,   // [12800][512]
                               const __bf16* __restrict__ WihT, // [3072][512]
                               const float* __restrict__ b_ih,
                               float* __restrict__ GI) {
    int lane = threadIdx.x & 31;
    int wave = threadIdx.x >> 5;
    int m0 = blockIdx.x * 64 + (wave & 1) * 32;
    int n0 = blockIdx.y * 128 + (wave >> 1) * 32;
    int rowA = m0 + (lane & 15);
    v8f c00 = {}, c01 = {}, c10 = {}, c11 = {};
    for (int k0 = 0; k0 < IN_; k0 += 32) {
        v16bf a0 = load_a_bf16(Xb, IN_, rowA,      k0, lane);
        v16bf a1 = load_a_bf16(Xb, IN_, rowA + 16, k0, lane);
        v16bf b0 = load_b_bf16(WihT, IN_, n0,      k0, lane);
        v16bf b1 = load_b_bf16(WihT, IN_, n0 + 16, k0, lane);
        c00 = wmma_bf16(a0, b0, c00);
        c01 = wmma_bf16(a0, b1, c01);
        c10 = wmma_bf16(a1, b0, c10);
        c11 = wmma_bf16(a1, b1, c11);
    }
    int hiM = (lane >> 4) << 3;
    int nn  = lane & 15;
#pragma unroll
    for (int nt = 0; nt < 2; ++nt) {
        int n = n0 + nt * 16 + nn;
        float bias = b_ih[n];
        v8f* cs[2] = {nt ? &c01 : &c00, nt ? &c11 : &c10};
#pragma unroll
        for (int mt = 0; mt < 2; ++mt) {
            int mB = m0 + mt * 16 + hiM;
            v8f c = *cs[mt];
#pragma unroll
            for (int r = 0; r < 8; ++r)
                GI[(long)(mB + r) * G3_ + n] = c[r] + bias;
        }
    }
}

// ---------------------------------------------------------------------------
// One recurrent step: gh = h @ mw_hh.T + b_hh for all 3 gates, fused with the
// GRU nonlinearity and h update.
// Block covers all M=128 x j-slab of 32 (weights read once from L2 per step).
// 8 waves = 4(m-groups of 32) x 2(j-tiles of 16); wave owns 2 m-tiles x 3
// gates = 6 accumulators, 6 WMMAs per 5 fragment loads.  grid (32), block 256.
// ---------------------------------------------------------------------------
__global__ void gru_step_kernel(const float* __restrict__ GI, int t,
                                const float* __restrict__ h_in,
                                const __bf16* __restrict__ h_in_bf,
                                float* __restrict__ h_out,
                                __bf16* __restrict__ h_out_bf,
                                const __bf16* __restrict__ Whh,   // [3072][1024]
                                const float* __restrict__ b_hh) {
    int lane = threadIdx.x & 31;
    int wave = threadIdx.x >> 5;
    int m0 = (wave & 3) * 32;                          // rows m0..m0+31
    int j0 = blockIdx.x * 32 + (wave >> 2) * 16;       // 16 j-columns
    int rowA = m0 + (lane & 15);
    v8f ar0 = {}, az0 = {}, an0 = {};
    v8f ar1 = {}, az1 = {}, an1 = {};
    for (int k0 = 0; k0 < H_; k0 += 32) {
        v16bf a0 = load_a_bf16(h_in_bf, H_, rowA,      k0, lane);
        v16bf a1 = load_a_bf16(h_in_bf, H_, rowA + 16, k0, lane);
        v16bf br = load_b_bf16(Whh, H_, j0,            k0, lane);
        v16bf bz = load_b_bf16(Whh, H_, H_ + j0,       k0, lane);
        v16bf bn = load_b_bf16(Whh, H_, 2 * H_ + j0,   k0, lane);
        ar0 = wmma_bf16(a0, br, ar0);
        az0 = wmma_bf16(a0, bz, az0);
        an0 = wmma_bf16(a0, bn, an0);
        ar1 = wmma_bf16(a1, br, ar1);
        az1 = wmma_bf16(a1, bz, az1);
        an1 = wmma_bf16(a1, bn, an1);
    }
    int j = j0 + (lane & 15);
    float bhr = b_hh[j], bhz = b_hh[H_ + j], bhn = b_hh[2 * H_ + j];
    int hiM = (lane >> 4) << 3;
#pragma unroll
    for (int mt = 0; mt < 2; ++mt) {
        v8f ar = mt ? ar1 : ar0;
        v8f az = mt ? az1 : az0;
        v8f an = mt ? an1 : an0;
        int mB = m0 + mt * 16 + hiM;
#pragma unroll
        for (int r = 0; r < 8; ++r) {
            int b = mB + r;
            const float* gi = GI + ((long)b * T_ + t) * G3_;
            float ghr = ar[r] + bhr;
            float ghz = az[r] + bhz;
            float ghn = an[r] + bhn;
            float rr = 1.0f / (1.0f + __expf(-(gi[j] + ghr)));
            float zz = 1.0f / (1.0f + __expf(-(gi[H_ + j] + ghz)));
            float nn = tanhf(gi[2 * H_ + j] + rr * ghn);
            float hp = h_in[(long)b * H_ + j];
            float hn = (1.0f - zz) * nn + zz * hp;
            h_out[(long)b * H_ + j]    = hn;
            h_out_bf[(long)b * H_ + j] = f2bf(hn);
        }
    }
}

// ---------------------------------------------------------------------------
// MoE layer: y[b,o] = sum_e omega[b,e] * ( (x @ w[e])[b,o] + bias[e,o] )
// mode 0: ELU -> bf16 out ; mode 1: clip[0,1] -> f32 out (final)
// grid (8, O/128), block 256.
// ---------------------------------------------------------------------------
__global__ void moe_kernel(const __bf16* __restrict__ Xb,      // [128][K]
                           const __bf16* __restrict__ WT,      // [E][O][K]
                           const float* __restrict__ bias,     // [E][O]
                           const float* __restrict__ omega,    // [128][8]
                           float* __restrict__ outF, __bf16* __restrict__ outB,
                           int K, int O, int mode) {
    int lane = threadIdx.x & 31;
    int wave = threadIdx.x >> 5;
    int m0 = blockIdx.x * 16;
    int o0 = blockIdx.y * 128 + wave * 16;
    int row = m0 + (lane & 15);
    int o   = o0 + (lane & 15);
    int mB  = m0 + ((lane >> 4) << 3);
    v8f tot = {};
    for (int e = 0; e < E_; ++e) {
        v8f acc = {};
        const __bf16* We = WT + (long)e * O * K;
        for (int k0 = 0; k0 < K; k0 += 32) {
            v16bf a = load_a_bf16(Xb, K, row, k0, lane);
            v16bf b = load_b_bf16(We, K, o0, k0, lane);
            acc = wmma_bf16(a, b, acc);
        }
        float be = bias[(long)e * O + o];
#pragma unroll
        for (int r = 0; r < 8; ++r) {
            float w = omega[(mB + r) * E_ + e];
            tot[r] += w * (acc[r] + be);
        }
    }
#pragma unroll
    for (int r = 0; r < 8; ++r) {
        int b = mB + r;
        float v = tot[r];
        if (mode == 0) {
            v = v > 0.0f ? v : (__expf(v) - 1.0f);       // ELU
            outB[(long)b * O + o] = f2bf(v);
        } else {
            v = fminf(fmaxf(v, 0.0f), 1.0f);             // clip
            outF[(long)b * O + o] = v;
        }
    }
}

// ---------------------------------------------------------------------------
// Gate GRUs (hidden=8, two layers) + softmax. Tiny -> one scalar workgroup.
// ---------------------------------------------------------------------------
__global__ void gate_gru_kernel(const float* __restrict__ X,
                                const float* __restrict__ wih0, const float* __restrict__ whh0,
                                const float* __restrict__ bih0, const float* __restrict__ bhh0,
                                const float* __restrict__ wih1, const float* __restrict__ whh1,
                                const float* __restrict__ bih1, const float* __restrict__ bhh1,
                                float* __restrict__ omega) {
    __shared__ float s_wih0[24 * 5], s_whh0[24 * 8], s_wih1[24 * 8], s_whh1[24 * 8];
    __shared__ float s_bih0[24], s_bhh0[24], s_bih1[24], s_bhh1[24];
    __shared__ float h0[B_ * 8], h1[B_ * 8];
    int tid = threadIdx.x;  // 256 threads
    for (int i = tid; i < 120; i += 256) s_wih0[i] = wih0[i];
    for (int i = tid; i < 192; i += 256) {
        s_whh0[i] = whh0[i]; s_wih1[i] = wih1[i]; s_whh1[i] = whh1[i];
    }
    for (int i = tid; i < 24; i += 256) {
        s_bih0[i] = bih0[i]; s_bhh0[i] = bhh0[i];
        s_bih1[i] = bih1[i]; s_bhh1[i] = bhh1[i];
    }
    for (int i = tid; i < B_ * 8; i += 256) { h0[i] = 0.0f; h1[i] = 0.0f; }
    __syncthreads();

    for (int t = 0; t < T_; ++t) {
        float n0v[4];
        // ---- layer 0 (input = X[:, t, 507:512]) ----
#pragma unroll
        for (int p = 0; p < 4; ++p) {
            int slot = tid * 4 + p;          // 1024 slots = 128 b x 8 u
            int b = slot >> 3, u = slot & 7;
            const float* x = X + ((long)b * T_ + t) * IN_ + (IN_ - 5);
            float gi[3], gh[3];
#pragma unroll
            for (int g = 0; g < 3; ++g) {
                int rI = g * 8 + u;
                float a = s_bih0[rI], c = s_bhh0[rI];
#pragma unroll
                for (int i = 0; i < 5; ++i) a += x[i] * s_wih0[rI * 5 + i];
#pragma unroll
                for (int j = 0; j < 8; ++j) c += h0[b * 8 + j] * s_whh0[rI * 8 + j];
                gi[g] = a; gh[g] = c;
            }
            float r = 1.0f / (1.0f + __expf(-(gi[0] + gh[0])));
            float z = 1.0f / (1.0f + __expf(-(gi[1] + gh[1])));
            float n = tanhf(gi[2] + r * gh[2]);
            n0v[p] = (1.0f - z) * n + z * h0[b * 8 + u];
        }
        __syncthreads();
#pragma unroll
        for (int p = 0; p < 4; ++p) h0[tid * 4 + p] = n0v[p];
        __syncthreads();
        // ---- layer 1 (input = new h0) ----
        float n1v[4];
#pragma unroll
        for (int p = 0; p < 4; ++p) {
            int slot = tid * 4 + p;
            int b = slot >> 3, u = slot & 7;
            float gi[3], gh[3];
#pragma unroll
            for (int g = 0; g < 3; ++g) {
                int rI = g * 8 + u;
                float a = s_bih1[rI], c = s_bhh1[rI];
#pragma unroll
                for (int j = 0; j < 8; ++j) {
                    a += h0[b * 8 + j] * s_wih1[rI * 8 + j];
                    c += h1[b * 8 + j] * s_whh1[rI * 8 + j];
                }
                gi[g] = a; gh[g] = c;
            }
            float r = 1.0f / (1.0f + __expf(-(gi[0] + gh[0])));
            float z = 1.0f / (1.0f + __expf(-(gi[1] + gh[1])));
            float n = tanhf(gi[2] + r * gh[2]);
            n1v[p] = (1.0f - z) * n + z * h1[b * 8 + u];
        }
        __syncthreads();
#pragma unroll
        for (int p = 0; p < 4; ++p) h1[tid * 4 + p] = n1v[p];
        __syncthreads();
    }
    // softmax over h1 rows
    if (tid < B_) {
        float m = -1e30f;
#pragma unroll
        for (int e = 0; e < 8; ++e) m = fmaxf(m, h1[tid * 8 + e]);
        float ex[8], s = 0.0f;
#pragma unroll
        for (int e = 0; e < 8; ++e) { ex[e] = __expf(h1[tid * 8 + e] - m); s += ex[e]; }
        float inv = 1.0f / s;
#pragma unroll
        for (int e = 0; e < 8; ++e) omega[tid * 8 + e] = ex[e] * inv;
    }
}

// ---------------------------------------------------------------------------
// Host launcher
// ---------------------------------------------------------------------------
extern "C" void kernel_launch(void* const* d_in, const int* in_sizes, int n_in,
                              void* d_out, int out_size, void* d_ws, size_t ws_size,
                              hipStream_t stream) {
    (void)in_sizes; (void)n_in; (void)out_size; (void)ws_size;
    const float* X      = (const float*)d_in[0];
    const float* gw_ih0 = (const float*)d_in[1];
    const float* gw_hh0 = (const float*)d_in[2];
    const float* gb_ih0 = (const float*)d_in[3];
    const float* gb_hh0 = (const float*)d_in[4];
    const float* gw_ih1 = (const float*)d_in[5];
    const float* gw_hh1 = (const float*)d_in[6];
    const float* gb_ih1 = (const float*)d_in[7];
    const float* gb_hh1 = (const float*)d_in[8];
    const float* mw_ih  = (const float*)d_in[9];
    const float* mw_hh  = (const float*)d_in[10];
    const float* mb_ih  = (const float*)d_in[11];
    const float* mb_hh  = (const float*)d_in[12];
    const float* w1     = (const float*)d_in[13];
    const float* b1     = (const float*)d_in[14];
    const float* w2     = (const float*)d_in[15];
    const float* b2     = (const float*)d_in[16];
    const float* w3     = (const float*)d_in[17];
    const float* b3     = (const float*)d_in[18];
    float* Y = (float*)d_out;

    // Workspace carve (all buffers rewritten every call -> deterministic)
    char* ws = (char*)d_ws;
    size_t off = 0;
    auto carve = [&](size_t bytes) {
        void* p = ws + off;
        off = (off + bytes + 255) & ~(size_t)255;
        return p;
    };
    float*  GI        = (float*) carve((size_t)M_ * G3_ * 4);       // 157 MB
    __bf16* Xbf       = (__bf16*)carve((size_t)M_ * IN_ * 2);       // 13 MB
    __bf16* mwih_bf   = (__bf16*)carve((size_t)G3_ * IN_ * 2);
    __bf16* mwhh_bf   = (__bf16*)carve((size_t)G3_ * H_ * 2);
    __bf16* w1T       = (__bf16*)carve((size_t)E_ * HH_ * H_ * 2);
    __bf16* w2T       = (__bf16*)carve((size_t)E_ * HH_ * HH_ * 2);
    __bf16* w3T       = (__bf16*)carve((size_t)E_ * HH_ * HH_ * 2);
    float*  hf0       = (float*) carve((size_t)B_ * H_ * 4);
    float*  hf1       = (float*) carve((size_t)B_ * H_ * 4);
    __bf16* hb0       = (__bf16*)carve((size_t)B_ * H_ * 2);
    __bf16* hb1       = (__bf16*)carve((size_t)B_ * H_ * 2);
    float*  omega     = (float*) carve((size_t)B_ * E_ * 4);
    __bf16* a1b       = (__bf16*)carve((size_t)B_ * HH_ * 2);
    __bf16* a2b       = (__bf16*)carve((size_t)B_ * HH_ * 2);

    // 1) packs: X + GEMM weights -> bf16; MoE weights transposed to [E][O][I]
    {
        long nx = (long)M_ * IN_;
        f32_to_bf16_kernel<<<(int)((nx + 255) / 256), 256, 0, stream>>>(X, Xbf, nx);
        long n1 = (long)G3_ * IN_;
        f32_to_bf16_kernel<<<(int)((n1 + 255) / 256), 256, 0, stream>>>(mw_ih, mwih_bf, n1);
        long n2 = (long)G3_ * H_;
        f32_to_bf16_kernel<<<(int)((n2 + 255) / 256), 256, 0, stream>>>(mw_hh, mwhh_bf, n2);
        long t1 = (long)E_ * H_ * HH_;
        transpose_moe_w_kernel<<<(int)((t1 + 255) / 256), 256, 0, stream>>>(w1, w1T, H_, HH_);
        long t2 = (long)E_ * HH_ * HH_;
        transpose_moe_w_kernel<<<(int)((t2 + 255) / 256), 256, 0, stream>>>(w2, w2T, HH_, HH_);
        transpose_moe_w_kernel<<<(int)((t2 + 255) / 256), 256, 0, stream>>>(w3, w3T, HH_, HH_);
    }

    // 2) GI = X @ mw_ih.T + mb_ih  (parallel, non-recurrent part of main GRU)
    gi_gemm_kernel<<<dim3(M_ / 64, G3_ / 128), 256, 0, stream>>>(Xbf, mwih_bf, mb_ih, GI);

    // 3) tiny gate GRUs + softmax -> omega
    gate_gru_kernel<<<1, 256, 0, stream>>>(X, gw_ih0, gw_hh0, gb_ih0, gb_hh0,
                                           gw_ih1, gw_hh1, gb_ih1, gb_hh1, omega);

    // 4) recurrent main GRU: 100 sequential fused-step launches (double-buffered)
    zero_h_kernel<<<(B_ * H_ + 255) / 256, 256, 0, stream>>>(hf0, hb0);
    float*  hf[2] = {hf0, hf1};
    __bf16* hb[2] = {hb0, hb1};
    for (int t = 0; t < T_; ++t) {
        int pi = t & 1;
        gru_step_kernel<<<dim3(H_ / 32), 256, 0, stream>>>(
            GI, t, hf[pi], hb[pi], hf[pi ^ 1], hb[pi ^ 1], mwhh_bf, mb_hh);
    }
    // T=100 (even) -> final hidden state lands back in buffer 0
    const __bf16* Zb = hb[0];

    // 5) MoE stack
    moe_kernel<<<dim3(B_ / 16, HH_ / 128), 256, 0, stream>>>(
        Zb, w1T, b1, omega, nullptr, a1b, H_, HH_, 0);
    moe_kernel<<<dim3(B_ / 16, HH_ / 128), 256, 0, stream>>>(
        a1b, w2T, b2, omega, nullptr, a2b, HH_, HH_, 0);
    moe_kernel<<<dim3(B_ / 16, HH_ / 128), 256, 0, stream>>>(
        a2b, w3T, b3, omega, Y, nullptr, HH_, HH_, 1);
}
